// PointConv_80066780332161
// MI455X (gfx1250) — matrix operand order repaired and checked by reading, not compiled, not run on hardware
//
#include <hip/hip_runtime.h>
#include <hip/hip_bf16.h>
#include <stdint.h>

// ---------------- problem constants (match reference) ----------------
#define BQ   16
#define NQ   8192
#define DQ   64
#define MQ   256
#define SQ   32
#define OUTQ 128
#define WCQ  64
#define RAD2 0.16f                      // RADIUS^2
#define BNS  0.9999950000374997f        // 1/sqrt(1 + 1e-5)

typedef __attribute__((ext_vector_type(16))) __bf16 v16bf;
typedef __attribute__((ext_vector_type(8)))  float  v8f;

// ---------------- fragment-layout index helpers (ISA 7.12.2, wave32) ------
// A 16x32 bf16 fragment: lane = (m&15) + 16*((k>>3)&1); slot below.
__device__ __forceinline__ int a_slot(int k) {
  int g  = (k >> 3) & 1;
  int kp = k - 8 * g;                    // in {0..7} U {16..23} (+1 for odd)
  int v  = (kp < 16) ? (kp >> 1) : (4 + ((kp - 16) >> 1));
  return 2 * v + (k & 1);
}
__device__ __forceinline__ int a_lane(int m, int k) {
  return (m & 15) + 16 * ((k >> 3) & 1);
}
// B 32x16 bf16 fragment: lane = n + 16*((k>>3)&1); slot = (k&7) + 8*(k>>4).
__device__ __forceinline__ int b_lane(int n, int k) { return n + 16 * ((k >> 3) & 1); }
__device__ __forceinline__ int b_slot(int k)        { return (k & 7) + 8 * (k >> 4); }

// LDS byte offset of a generic pointer to __shared__ memory (low 32 bits of
// the shared aperture address are the LDS offset).
__device__ __forceinline__ unsigned lds_off(const void* p) {
  return (unsigned)(unsigned long long)
      (__attribute__((address_space(3))) const void*)p;
}

// ---------------- 1) Farthest point sampling ------------------------------
// One block per batch. The whole 96KB point cloud is pulled into LDS once via
// CDNA5 async global->LDS copies (ASYNCcnt), then 255 serial argmax rounds
// run entirely out of LDS.
__global__ void fps_kernel(const float* __restrict__ xyz,
                           float* __restrict__ new_xyz) {
  extern __shared__ __align__(16) unsigned char dynsmem[];
  float* sX = (float*)dynsmem;                       // NQ*3 floats = 96KB
  const int b = blockIdx.x;
  const int t = threadIdx.x;
  const float* X = xyz + (size_t)b * NQ * 3;

  // async copy: 98304 bytes = 256 threads * 24 * 16B
  const unsigned ldsbase = lds_off(sX);
  const unsigned long long gbase = (unsigned long long)(size_t)X;
#pragma unroll
  for (int q = 0; q < (NQ * 3 * 4) / (256 * 16); ++q) {
    unsigned loff = ldsbase + (unsigned)((t + 256 * q) * 16);
    unsigned long long g = gbase + (unsigned long long)((t + 256 * q) * 16);
    asm volatile("global_load_async_to_lds_b128 %0, %1, off"
                 :: "v"(loff), "v"(g) : "memory");
  }
  asm volatile("s_wait_asynccnt 0x0" ::: "memory");
  __syncthreads();

  float dist[NQ / 256];
#pragma unroll
  for (int j = 0; j < NQ / 256; ++j) dist[j] = 1e10f;

  __shared__ float rv[256];
  __shared__ int   ri[256];
  __shared__ int   slast;
  __shared__ float spt[3];

  if (t == 0) {
    slast = 0;
    new_xyz[(size_t)(b * MQ + 0) * 3 + 0] = sX[0];
    new_xyz[(size_t)(b * MQ + 0) * 3 + 1] = sX[1];
    new_xyz[(size_t)(b * MQ + 0) * 3 + 2] = sX[2];
  }
  __syncthreads();

  for (int it = 1; it < MQ; ++it) {
    if (t == 0) {
      int last = slast;
      spt[0] = sX[last * 3 + 0];
      spt[1] = sX[last * 3 + 1];
      spt[2] = sX[last * 3 + 2];
    }
    __syncthreads();
    const float px = spt[0], py = spt[1], pz = spt[2];

    float bv = -1.0f;
    int   bi = 0;
#pragma unroll
    for (int j = 0; j < NQ / 256; ++j) {
      const int n = t + 256 * j;
      float dx = sX[n * 3 + 0] - px;
      float dy = sX[n * 3 + 1] - py;
      float dz = sX[n * 3 + 2] - pz;
      float d  = dx * dx + dy * dy + dz * dz;
      float nd = fminf(dist[j], d);
      dist[j]  = nd;
      if (nd > bv) { bv = nd; bi = n; }   // strict > keeps lowest index on tie
    }
    rv[t] = bv; ri[t] = bi;
    __syncthreads();
    for (int s = 128; s > 0; s >>= 1) {
      if (t < s) {
        if (rv[t + s] > rv[t] || (rv[t + s] == rv[t] && ri[t + s] < ri[t])) {
          rv[t] = rv[t + s]; ri[t] = ri[t + s];
        }
      }
      __syncthreads();
    }
    if (t == 0) {
      int w = ri[0];
      slast = w;
      new_xyz[(size_t)(b * MQ + it) * 3 + 0] = sX[w * 3 + 0];
      new_xyz[(size_t)(b * MQ + it) * 3 + 1] = sX[w * 3 + 1];
      new_xyz[(size_t)(b * MQ + it) * 3 + 2] = sX[w * 3 + 2];
    }
    __syncthreads();
  }
}

// ---------------- 2) Ball query (first S in-radius indices, wave per row) --
__global__ void ballq_kernel(const float* __restrict__ xyz,
                             const float* __restrict__ new_xyz,
                             int* __restrict__ gidx) {
  const int wid  = threadIdx.x >> 5;
  const int lane = threadIdx.x & 31;
  const int row  = blockIdx.x * 8 + wid;             // 0 .. B*M-1
  const int b = row / MQ;
  const float* X = xyz + (size_t)b * NQ * 3;
  const float cx = new_xyz[(size_t)row * 3 + 0];
  const float cy = new_xyz[(size_t)row * 3 + 1];
  const float cz = new_xyz[(size_t)row * 3 + 2];
  int* out = gidx + (size_t)row * SQ;

  int cnt = 0;
  int first = 0;
  for (int base = 0; base < NQ; base += 32) {
    const int n = base + lane;
    float dx = X[n * 3 + 0] - cx;
    float dy = X[n * 3 + 1] - cy;
    float dz = X[n * 3 + 2] - cz;
    float d2 = dx * dx + dy * dy + dz * dz;
    const bool in = (d2 <= RAD2);
    unsigned mask = (unsigned)__ballot(in);          // wave32: low 32 bits
    if (cnt == 0 && mask) first = base + (__ffs((int)mask) - 1);
    if (in) {
      int pos = cnt + __popc(mask & ((1u << lane) - 1u));
      if (pos < SQ) out[pos] = n;
    }
    cnt += __popc(mask);
    if (cnt >= SQ) break;
  }
  if (cnt < SQ) {
    const int fill = (cnt > 0) ? first : 0;
    for (int j = cnt + lane; j < SQ; j += 32) out[j] = fill;
  }
}

// ---------------- 3) pre-swizzle shared weight matrices to A-frag layout ---
// wl (128 x 4096) -> wlf [kc=128][tile=8][lane=32][slot=16]
// wf (128 x 192)  -> wff [kc=6]  [tile=8][lane=32][slot=16]
__global__ void prep_frag_kernel(const float* __restrict__ wl,
                                 const float* __restrict__ wf,
                                 __bf16* __restrict__ wlf,
                                 __bf16* __restrict__ wff) {
  const int blk = blockIdx.x;
  const float* src; __bf16* dst; int K; int kc;
  if (blk < 128) { src = wl; dst = wlf; K = DQ * WCQ; kc = blk; }
  else           { src = wf; dst = wff; K = DQ + OUTQ; kc = blk - 128; }
  for (int i = threadIdx.x; i < OUTQ * 32; i += 256) {
    int o = i >> 5, k = i & 31;
    int tile = o >> 4;
    size_t di = (((size_t)kc * 8 + tile) * 32 + a_lane(o, k)) * 16 + a_slot(k);
    dst[di] = (__bf16)src[(size_t)o * K + kc * 32 + k];
  }
}

// ---------------- 4) group + 3-layer MLP + agg via WMMA --------------------
// One block per (b,m). agg[b,m,d,w] = sum_s feat[d,s] * weights[w,s], written
// straight to global in B-fragment layout for the wl GEMM:
//   aggf [b][m>>4][kc=128][lane=32][slot=16]   (n-col inside tile = m&15)
__global__ void groupmlp_kernel(const float* __restrict__ xyz,
                                const float* __restrict__ feature,
                                const float* __restrict__ new_xyz,
                                const int* __restrict__ gidx,
                                const float* __restrict__ w0, const float* __restrict__ b0,
                                const float* __restrict__ g0, const float* __restrict__ be0,
                                const float* __restrict__ w1, const float* __restrict__ b1,
                                const float* __restrict__ g1, const float* __restrict__ be1,
                                const float* __restrict__ w2, const float* __restrict__ b2,
                                __bf16* __restrict__ aggf) {
  const int m = blockIdx.x, b = blockIdx.y;
  const int t = threadIdx.x;
  const int row = b * MQ + m;

  __shared__ int    sg[SQ];
  __shared__ float  cx[3][SQ];
  __shared__ float  h0[WCQ][SQ];
  __shared__ float  h1[WCQ][SQ];
  __shared__ __align__(32) __bf16 fragA[4][32][16];  // feat: 4 m-tiles (d)
  __shared__ __align__(32) __bf16 fragB[4][32][16];  // weightsT: 4 n-tiles (w)

  if (t < SQ) sg[t] = gidx[(size_t)row * SQ + t];
  __syncthreads();
  if (t < 96) {
    int c = t / SQ, s = t % SQ;
    cx[c][s] = xyz[((size_t)b * NQ + sg[s]) * 3 + c] - new_xyz[(size_t)row * 3 + c];
  }
  __syncthreads();

  for (int i = t; i < WCQ * SQ; i += 256) {          // layer 0 (K=3) + BN + relu
    int o = i >> 5, s = i & 31;
    float acc = b0[o];
#pragma unroll
    for (int c = 0; c < 3; ++c) acc += w0[o * 3 + c] * cx[c][s];
    float y = g0[o] * (acc * BNS) + be0[o];
    h0[o][s] = y > 0.f ? y : 0.f;
  }
  __syncthreads();
  for (int i = t; i < WCQ * SQ; i += 256) {          // layer 1 (K=64) + BN + relu
    int o = i >> 5, s = i & 31;
    float acc = b1[o];
    for (int c = 0; c < WCQ; ++c) acc += w1[o * WCQ + c] * h0[c][s];
    float y = g1[o] * (acc * BNS) + be1[o];
    h1[o][s] = y > 0.f ? y : 0.f;
  }
  __syncthreads();
  for (int i = t; i < WCQ * SQ; i += 256) {          // weights = w2@h1 + b2 -> B frag
    int o = i >> 5, s = i & 31;
    float acc = b2[o];
    for (int c = 0; c < WCQ; ++c) acc += w2[o * WCQ + c] * h1[c][s];
    fragB[o >> 4][b_lane(o & 15, s)][b_slot(s)] = (__bf16)acc;
  }
  for (int i = t; i < DQ * SQ; i += 256) {           // grouped features -> A frag
    int d = i >> 5, s = i & 31;
    fragA[d >> 4][a_lane(d, s)][a_slot(s)] =
        (__bf16)feature[((size_t)b * DQ + d) * NQ + sg[s]];
  }
  __syncthreads();

  const int wid = t >> 5, lane = t & 31;
  const int ncol = m & 15, mtile = m >> 4;
  const size_t bbase = ((size_t)b * (MQ / 16) + mtile) * 128;   // * kc dim
#pragma unroll
  for (int q = 0; q < 2; ++q) {                      // 16 tiles / 8 waves
    int ti = wid * 2 + q;
    int mt = ti >> 2, nt = ti & 3;
    v16bf a  = *(const v16bf*)(&fragA[mt][lane][0]);
    v16bf bm = *(const v16bf*)(&fragB[nt][lane][0]);
    v8f   c  = {};
    c = __builtin_amdgcn_wmma_f32_16x16x32_bf16(false, a, false, bm, (short)0, c,
                                                false, false);
    // scatter C into aggf B-fragment layout: k = dd*64 + ww
    int ww   = nt * 16 + (lane & 15);
    int k5   = ww & 31;
    int bl_  = b_lane(ncol, k5);
    int sl_  = b_slot(k5);
    int kcof = (ww >> 5);
#pragma unroll
    for (int r = 0; r < 8; ++r) {
      int dd = mt * 16 + r + 8 * (lane >> 4);
      int kc = dd * 2 + kcof;
      aggf[((bbase + kc) * 32 + bl_) * 16 + sl_] = (__bf16)c[r];
    }
  }
}

// ---------------- 5) new_feature = relu(bn(wl @ agg + bl)) via WMMA -------
// Block: (b, 16-column m-tile); 8 waves = 8 o-tiles. Inner loop is pure
// fragment streaming: 4x global_load_b128 + 1 WMMA per k-step, no LDS.
__global__ void linl_kernel(const __bf16* __restrict__ wlf,
                            const __bf16* __restrict__ aggf,
                            const float* __restrict__ bl,
                            const float* __restrict__ gl, const float* __restrict__ bel,
                            float* __restrict__ nf) {
  const int mt = blockIdx.x, b = blockIdx.y;
  const int t = threadIdx.x, wid = t >> 5, lane = t & 31;

  const __bf16* pa = wlf + ((size_t)wid * 32 + lane) * 16;
  const __bf16* pb = aggf + (((size_t)b * (MQ / 16) + mt) * 128 * 32 + lane) * 16;

  v8f acc = {};
  for (int kc = 0; kc < (DQ * WCQ) / 32; ++kc) {     // 128 k-steps
    v16bf a  = *(const v16bf*)(pa + (size_t)kc * (8 * 32 * 16));
    v16bf bm = *(const v16bf*)(pb + (size_t)kc * (32 * 16));
    if (kc + 4 < (DQ * WCQ) / 32)                    // global_prefetch_b8 path
      __builtin_prefetch((const void*)(pb + (size_t)(kc + 4) * (32 * 16)), 0, 0);
    acc = __builtin_amdgcn_wmma_f32_16x16x32_bf16(false, a, false, bm, (short)0, acc,
                                                  false, false);
  }
#pragma unroll
  for (int r = 0; r < 8; ++r) {
    int o    = wid * 16 + r + 8 * (lane >> 4);
    int mcol = mt * 16 + (lane & 15);
    float x = acc[r] + bl[o];
    float y = gl[o] * (x * BNS) + bel[o];
    nf[((size_t)b * OUTQ + o) * MQ + mcol] = y > 0.f ? y : 0.f;
  }
}

// ---------------- 6) 3-NN of xyz against new_xyz + IDW weights ------------
__global__ void knn_kernel(const float* __restrict__ xyz,
                           const float* __restrict__ new_xyz,
                           int* __restrict__ knni, float* __restrict__ knnw) {
  const int b = blockIdx.y;
  const int n = blockIdx.x * 256 + threadIdx.x;
  __shared__ float sx[MQ * 3];
  for (int i = threadIdx.x; i < MQ * 3; i += 256)
    sx[i] = new_xyz[(size_t)b * MQ * 3 + i];
  __syncthreads();

  const float px = xyz[((size_t)b * NQ + n) * 3 + 0];
  const float py = xyz[((size_t)b * NQ + n) * 3 + 1];
  const float pz = xyz[((size_t)b * NQ + n) * 3 + 2];
  float d0 = 3e38f, d1 = 3e38f, d2v = 3e38f;
  int   i0 = 0, i1 = 0, i2 = 0;
  for (int m = 0; m < MQ; ++m) {
    float dx = px - sx[m * 3 + 0];
    float dy = py - sx[m * 3 + 1];
    float dz = pz - sx[m * 3 + 2];
    float d  = dx * dx + dy * dy + dz * dz;
    if (d < d0)       { d2v = d1; i2 = i1; d1 = d0; i1 = i0; d0 = d; i0 = m; }
    else if (d < d1)  { d2v = d1; i2 = i1; d1 = d;  i1 = m; }
    else if (d < d2v) { d2v = d;  i2 = m; }
  }
  float r0 = 1.f / (d0 + 1e-8f), r1 = 1.f / (d1 + 1e-8f), r2 = 1.f / (d2v + 1e-8f);
  float s = r0 + r1 + r2;
  size_t o = ((size_t)b * NQ + n) * 3;
  knni[o] = i0; knni[o + 1] = i1; knni[o + 2] = i2;
  knnw[o] = r0 / s; knnw[o + 1] = r1 / s; knnw[o + 2] = r2 / s;
}

// ---------------- 7) interpolate + concat + final conv via WMMA -----------
// Block: (b, 64-column n-tile). cat = [interp(128); feature(64)] built in
// B-fragment layout in LDS; A streamed from pre-swizzled wff.
__global__ void final_kernel(const float* __restrict__ feature,
                             const float* __restrict__ nf,
                             const int* __restrict__ knni, const float* __restrict__ knnw,
                             const __bf16* __restrict__ wff,
                             const float* __restrict__ bfv,
                             const float* __restrict__ gf, const float* __restrict__ bef,
                             float* __restrict__ out) {
  const int b  = blockIdx.y;
  const int n0 = blockIdx.x * 64;
  const int t = threadIdx.x, wid = t >> 5, lane = t & 31;
  __shared__ __align__(32) __bf16 fragC[6][4][32][16];   // [kc][nt][lane][slot]

  for (int i = t; i < OUTQ * 64; i += 256) {       // interp rows (K = 0..127)
    int c = i >> 6, j = i & 63;
    size_t nn = (size_t)b * NQ + n0 + j;
    const int*   ki = &knni[nn * 3];
    const float* kw = &knnw[nn * 3];
    const float* nfb = nf + ((size_t)b * OUTQ + c) * MQ;
    float v = kw[0] * nfb[ki[0]] + kw[1] * nfb[ki[1]] + kw[2] * nfb[ki[2]];
    fragC[c >> 5][j >> 4][b_lane(j & 15, c & 31)][b_slot(c & 31)] = (__bf16)v;
  }
  for (int i = t; i < DQ * 64; i += 256) {         // feature rows (K = 128..191)
    int cf = i >> 6, j = i & 63;
    int c = OUTQ + cf;
    fragC[c >> 5][j >> 4][b_lane(j & 15, c & 31)][b_slot(c & 31)] =
        (__bf16)feature[((size_t)b * DQ + cf) * NQ + n0 + j];
  }
  __syncthreads();

  const __bf16* pa = wff + ((size_t)wid * 32 + lane) * 16;
#pragma unroll
  for (int nt = 0; nt < 4; ++nt) {
    v8f acc = {};
#pragma unroll
    for (int kc = 0; kc < 6; ++kc) {
      v16bf a  = *(const v16bf*)(pa + (size_t)kc * (8 * 32 * 16));
      v16bf bm = *(const v16bf*)(&fragC[kc][nt][lane][0]);
      acc = __builtin_amdgcn_wmma_f32_16x16x32_bf16(false, a, false, bm, (short)0, acc,
                                                    false, false);
    }
#pragma unroll
    for (int r = 0; r < 8; ++r) {
      int o = wid * 16 + r + 8 * (lane >> 4);
      int n = n0 + nt * 16 + (lane & 15);
      float x = acc[r] + bfv[o];
      float y = gf[o] * (x * BNS) + bef[o];
      out[((size_t)b * OUTQ + o) * NQ + n] = y > 0.f ? y : 0.f;
    }
  }
}

// ---------------- launch ---------------------------------------------------
extern "C" void kernel_launch(void* const* d_in, const int* in_sizes, int n_in,
                              void* d_out, int out_size, void* d_ws, size_t ws_size,
                              hipStream_t stream) {
  (void)in_sizes; (void)n_in; (void)out_size; (void)ws_size;
  const float* xyz     = (const float*)d_in[0];
  const float* feature = (const float*)d_in[1];
  const float* w0 = (const float*)d_in[2];
  const float* b0 = (const float*)d_in[3];
  const float* g0 = (const float*)d_in[4];
  const float* be0 = (const float*)d_in[5];
  const float* w1 = (const float*)d_in[6];
  const float* b1 = (const float*)d_in[7];
  const float* g1 = (const float*)d_in[8];
  const float* be1 = (const float*)d_in[9];
  const float* w2 = (const float*)d_in[10];
  const float* b2 = (const float*)d_in[11];
  const float* wl = (const float*)d_in[12];
  const float* bl = (const float*)d_in[13];
  const float* gl = (const float*)d_in[14];
  const float* bel = (const float*)d_in[15];
  const float* wf = (const float*)d_in[16];
  const float* bf = (const float*)d_in[17];
  const float* gf = (const float*)d_in[18];
  const float* bef = (const float*)d_in[19];
  float* out = (float*)d_out;

  // workspace carve-up (~40.5 MB total, all offsets 32B-aligned)
  char* ws = (char*)d_ws;
  float* new_xyz = (float*)ws;    ws += (size_t)BQ * MQ * 3 * sizeof(float);
  int*   gidx    = (int*)ws;      ws += (size_t)BQ * MQ * SQ * sizeof(int);
  __bf16* aggf   = (__bf16*)ws;   ws += (size_t)BQ * MQ * DQ * WCQ * sizeof(__bf16);
  float* nf      = (float*)ws;    ws += (size_t)BQ * OUTQ * MQ * sizeof(float);
  int*   knni    = (int*)ws;      ws += (size_t)BQ * NQ * 3 * sizeof(int);
  float* knnw    = (float*)ws;    ws += (size_t)BQ * NQ * 3 * sizeof(float);
  __bf16* wlf    = (__bf16*)ws;   ws += (size_t)128 * 8 * 32 * 16 * sizeof(__bf16);
  __bf16* wff    = (__bf16*)ws;

  prep_frag_kernel<<<134, 256, 0, stream>>>(wl, wf, wlf, wff);
  fps_kernel<<<BQ, 256, NQ * 3 * sizeof(float), stream>>>(xyz, new_xyz);
  ballq_kernel<<<(BQ * MQ) / 8, 256, 0, stream>>>(xyz, new_xyz, gidx);
  groupmlp_kernel<<<dim3(MQ, BQ), 256, 0, stream>>>(
      xyz, feature, new_xyz, gidx, w0, b0, g0, be0, w1, b1, g1, be1, w2, b2, aggf);
  linl_kernel<<<dim3(MQ / 16, BQ), 256, 0, stream>>>(wlf, aggf, bl, gl, bel, nf);
  knn_kernel<<<dim3(NQ / 256, BQ), 256, 0, stream>>>(xyz, new_xyz, knni, knnw);
  final_kernel<<<dim3(NQ / 64, BQ), 256, 0, stream>>>(
      feature, nf, knni, knnw, wff, bf, gf, bef, out);
}
